// SelfAttention_72292889526352
// MI455X (gfx1250) — matrix operand order repaired and checked by reading, not compiled, not run on hardware
//
#include <hip/hip_runtime.h>
#include <hip/hip_bf16.h>

// ---------------------------------------------------------------------------
// CDNA5 (gfx1250) fused self-attention forward.
//   B=2, T=2048, C=1024, H=16, D=64.  wave32, WMMA bf16 16x16x32, f32 accum.
//   K tiles staged via Tensor Data Mover, mask tiles via async LDS loads.
// ---------------------------------------------------------------------------

typedef __bf16 bf16;
typedef __bf16 v16bf __attribute__((ext_vector_type(16)));
typedef __bf16 v8bf  __attribute__((ext_vector_type(8)));
typedef float  v8f   __attribute__((ext_vector_type(8)));
typedef unsigned int u32x4 __attribute__((ext_vector_type(4)));
typedef int          i32x4 __attribute__((ext_vector_type(4)));
typedef int          i32x8 __attribute__((ext_vector_type(8)));
typedef int          v4i_vs __attribute__((vector_size(16)));  // matches builtin proto

#if __has_builtin(__builtin_amdgcn_tensor_load_to_lds) && \
    __has_builtin(__builtin_amdgcn_s_wait_tensorcnt)
#define HAVE_TDM 1
#else
#define HAVE_TDM 0
#endif

#if __has_builtin(__builtin_amdgcn_global_load_async_to_lds_b128) && \
    __has_builtin(__builtin_amdgcn_s_wait_asynccnt)
#define HAVE_ASYNC 1
#else
#define HAVE_ASYNC 0
#endif

// async b128 builtin wants:  (AS1 v4i* src, AS3 v4i* dst, int offset, int cpol)
#define GV4(p) ((__attribute__((address_space(1))) v4i_vs*)(p))
#define LV4(p) ((__attribute__((address_space(3))) v4i_vs*)(p))

#define WMMA_BF16(a, b, c)                                                    \
    __builtin_amdgcn_wmma_f32_16x16x32_bf16(false, (a), false, (b),           \
                                            (short)0, (c), false, false)

union AFrag {
    v16bf v;
    v8bf  h[2];
};

__device__ __forceinline__ v8f vzero8() {
    v8f z;
#pragma unroll
    for (int i = 0; i < 8; ++i) z[i] = 0.0f;
    return z;
}

__device__ __forceinline__ float hmax16(float v) {
#pragma unroll
    for (int m = 1; m < 16; m <<= 1) v = fmaxf(v, __shfl_xor(v, m, 32));
    return v;
}
__device__ __forceinline__ float hsum16(float v) {
#pragma unroll
    for (int m = 1; m < 16; m <<= 1) v += __shfl_xor(v, m, 32);
    return v;
}

static constexpr int Bc = 2;
static constexpr int Tc = 2048;
static constexpr int Cc = 1024;
static constexpr int Hc = 16;
static constexpr int Dc = 64;

// ---------------------------------------------------------------------------
// Kernel 1: QKV projection.  out[b,h,t,d] = bf16( x[b,t,:] . W[h*64+d,:] + b )
// ---------------------------------------------------------------------------
__global__ __launch_bounds__(128) void qkv_proj_kernel(
    const float* __restrict__ x,
    const float* __restrict__ Wq, const float* __restrict__ bq,
    const float* __restrict__ Wk, const float* __restrict__ bk,
    const float* __restrict__ Wv, const float* __restrict__ bv,
    bf16* __restrict__ qb, bf16* __restrict__ kb, bf16* __restrict__ vb) {
    __shared__ bf16 lA[64][32];  // x tile  [m][k]
    __shared__ bf16 lB[64][32];  // W tile  [n][k]

    const int mblk  = blockIdx.x * 64;
    const int nblk  = blockIdx.y * 64;
    const int which = blockIdx.z;
    const float* W    = (which == 0) ? Wq : (which == 1) ? Wk : Wv;
    const float* bias = (which == 0) ? bq : (which == 1) ? bk : bv;
    bf16*        out  = (which == 0) ? qb : (which == 1) ? kb : vb;

    const int tid  = threadIdx.x;
    const int wave = tid >> 5;
    const int lane = tid & 31;
    const int half = lane >> 4;
    const int lmod = lane & 15;

    v8f acc[4];
#pragma unroll
    for (int i = 0; i < 4; ++i) acc[i] = vzero8();

    const int row  = tid >> 1;
    const int kseg = (tid & 1) * 16;

    for (int k0 = 0; k0 < Cc; k0 += 32) {
        {
            const float* xs = x + (size_t)(mblk + row) * Cc + k0 + kseg;
            const float* ws = W + (size_t)(nblk + row) * Cc + k0 + kseg;
#pragma unroll
            for (int j = 0; j < 16; ++j) lA[row][kseg + j] = (bf16)xs[j];
#pragma unroll
            for (int j = 0; j < 16; ++j) lB[row][kseg + j] = (bf16)ws[j];
        }
        __syncthreads();

        AFrag a;
        a.h[0] = *(const v8bf*)&lA[wave * 16 + lmod][half * 16];
        a.h[1] = *(const v8bf*)&lA[wave * 16 + lmod][half * 16 + 8];
#pragma unroll
        for (int nt = 0; nt < 4; ++nt) {
            AFrag b;
            b.v = *(const v16bf*)&lB[nt * 16 + lmod][half * 16];
            acc[nt] = WMMA_BF16(a.v, b.v, acc[nt]);
        }
        __syncthreads();
    }

#pragma unroll
    for (int nt = 0; nt < 4; ++nt) {
        const int   n   = nblk + nt * 16 + lmod;
        const float bv_ = bias[n];
        const int   h   = n >> 6;
        const int   d   = n & 63;
#pragma unroll
        for (int r = 0; r < 8; ++r) {
            const int m = mblk + wave * 16 + half * 8 + r;
            const int b = m >> 11;
            const int t = m & (Tc - 1);
            out[(((size_t)b * Hc + h) * Tc + t) * Dc + d] = (bf16)(acc[nt][r] + bv_);
        }
    }
}

// ---------------------------------------------------------------------------
// Kernel 2: flash-style masked attention.  Grid: (T/64, B*H), block = 128.
// ---------------------------------------------------------------------------
__global__ __launch_bounds__(128) void attn_kernel(
    const bf16* __restrict__ qb, const bf16* __restrict__ kb,
    const bf16* __restrict__ vb, const unsigned char* __restrict__ cmask,
    const unsigned char* __restrict__ dmask, bf16* __restrict__ yb) {
    __shared__ bf16 lK[64][64];     // [kv][d]
    __shared__ bf16 lV[64][64];     // [d][kv]  (transposed V)
    __shared__ bf16 lP[4][16][64];  // per-wave P tile [m][kv]
#if HAVE_ASYNC
    __shared__ unsigned char lCM[64][64];  // causal mask tile  [q][kv]
    __shared__ unsigned char lDM[64][64];  // dynamic mask tile [q][kv]
#endif

    const int qblk = blockIdx.x;
    const int bh   = blockIdx.y;
    const int b    = bh / Hc;
    const int h    = bh % Hc;

    const int tid  = threadIdx.x;
    const int wave = tid >> 5;
    const int lane = tid & 31;
    const int half = lane >> 4;
    const int lmod = lane & 15;

    const bf16* qh = qb + (size_t)bh * Tc * Dc;
    const bf16* kh = kb + (size_t)bh * Tc * Dc;
    const bf16* vh = vb + (size_t)bh * Tc * Dc;

    AFrag qf[2];
    {
        const bf16* qrow = qh + (size_t)(qblk * 64 + wave * 16 + lmod) * Dc;
        qf[0].h[0] = *(const v8bf*)(qrow + half * 16);
        qf[0].h[1] = *(const v8bf*)(qrow + half * 16 + 8);
        qf[1].h[0] = *(const v8bf*)(qrow + 32 + half * 16);
        qf[1].h[1] = *(const v8bf*)(qrow + 32 + half * 16 + 8);
    }

    v8f oacc[4];
#pragma unroll
    for (int i = 0; i < 4; ++i) oacc[i] = vzero8();
    float mrow[8], lrow[8];
#pragma unroll
    for (int r = 0; r < 8; ++r) { mrow[r] = -INFINITY; lrow[r] = 0.0f; }

    const int row = tid >> 1;        // 0..63
    const int seg = (tid & 1) * 32;  // 0 or 32
    const int nkv = qblk + 1;        // chunk structure bounds visibility
    const float scale = 0.125f;      // D^-0.5

    for (int kt = 0; kt < nkv; ++kt) {
        const int kbase = kt * 64;

        // ---- stage K tile (contiguous 8KB) via Tensor Data Mover ----------
#if HAVE_TDM
        if (wave == 0) {
            const unsigned long long ga =
                (unsigned long long)(uintptr_t)(kh + (size_t)kbase * Dc);
            const unsigned lds_addr =
                (unsigned)(uintptr_t)(__attribute__((address_space(3))) bf16*)&lK[0][0];
            u32x4 g0;
            g0[0] = 1u;                                  // count=1, user mode
            g0[1] = lds_addr;                            // D#.lds_addr
            g0[2] = (unsigned)(ga & 0xFFFFFFFFu);        // global_addr[31:0]
            g0[3] = (unsigned)((ga >> 32) & 0x01FFFFFFu) // global_addr[56:32]
                    | (2u << 30);                        // type = 2 ("image")
            i32x8 g1;
            g1[0] = (3 << 16);           // data_size = 8B, no mask/barrier/pad
            g1[1] = (int)(1024u << 16);  // tensor_dim0 = 1024 x 8B (lo 16b)
            g1[2] = (1 << 16);           // tensor_dim0 hi = 0, tensor_dim1 = 1
            g1[3] = (int)(1024u << 16);  // tile_dim0 = 1024
            g1[4] = 1;                   // tile_dim1 = 1, tile_dim2 = 0
            g1[5] = 1024;                // tensor_dim0_stride = 1024
            g1[6] = 0;
            g1[7] = 0;
            i32x4 z4 = {0, 0, 0, 0};
#if __clang_major__ >= 23
            i32x8 z8 = {0, 0, 0, 0, 0, 0, 0, 0};
            __builtin_amdgcn_tensor_load_to_lds(g0, g1, z4, z4, z8, 0);
#else
            __builtin_amdgcn_tensor_load_to_lds(g0, g1, z4, z4, 0);
#endif
            __builtin_amdgcn_s_wait_tensorcnt(0);
        }
#else
        {
            const bf16* ks = kh + (size_t)(kbase + row) * Dc + seg;
#pragma unroll
            for (int j = 0; j < 32; ++j) lK[row][seg + j] = ks[j];
        }
#endif

        // ---- stage V tile transposed (manual: transpose needs lane scatter)
        {
            const bf16* vs = vh + (size_t)(kbase + row) * Dc + seg;
#pragma unroll
            for (int j = 0; j < 32; ++j) lV[seg + j][row] = vs[j];
            if (kt + 1 < nkv)
                __builtin_prefetch(vh + (size_t)(kbase + 64 + row) * Dc, 0, 0);
        }

        // ---- stage mask tiles via async global->LDS copies ----------------
#if HAVE_ASYNC
        {
            const int moff = (tid & 1) * 32;
            const unsigned char* cs =
                cmask + (size_t)(qblk * 64 + row) * Tc + kbase + moff;
            const unsigned char* dsrc =
                dmask + ((size_t)b * Tc + qblk * 64 + row) * Tc + kbase + moff;
            unsigned char* lc = &lCM[row][moff];
            unsigned char* ld = &lDM[row][moff];
            __builtin_amdgcn_global_load_async_to_lds_b128(GV4(cs), LV4(lc), 0, 0);
            __builtin_amdgcn_global_load_async_to_lds_b128(GV4(cs + 16), LV4(lc + 16), 0, 0);
            __builtin_amdgcn_global_load_async_to_lds_b128(GV4(dsrc), LV4(ld), 0, 0);
            __builtin_amdgcn_global_load_async_to_lds_b128(GV4(dsrc + 16), LV4(ld + 16), 0, 0);
            __builtin_amdgcn_s_wait_asynccnt(0);
        }
#endif
        __syncthreads();

        // ---- S = Q K^T for four 16-wide kv sub-tiles ----------------------
        v8f sac[4];
#pragma unroll
        for (int nn = 0; nn < 4; ++nn) {
            v8f s = vzero8();
            AFrag b0, b1;
            b0.v = *(const v16bf*)&lK[nn * 16 + lmod][half * 16];
            b1.v = *(const v16bf*)&lK[nn * 16 + lmod][32 + half * 16];
            s = WMMA_BF16(qf[0].v, b0.v, s);
            s = WMMA_BF16(qf[1].v, b1.v, s);
            sac[nn] = s;
        }

        // ---- scale + masks ------------------------------------------------
#pragma unroll
        for (int nn = 0; nn < 4; ++nn) {
            const int kvl = nn * 16 + lmod;
#pragma unroll
            for (int r = 0; r < 8; ++r) {
                const int ql = wave * 16 + half * 8 + r;
#if HAVE_ASYNC
                const unsigned char cm = lCM[ql][kvl];
                const unsigned char dm = lDM[ql][kvl];
#else
                const int q  = qblk * 64 + ql;
                const int kv = kbase + kvl;
                const unsigned char cm = cmask[(size_t)q * Tc + kv];
                const unsigned char dm = dmask[((size_t)b * Tc + q) * Tc + kv];
#endif
                sac[nn][r] = (cm | dm) ? -INFINITY : sac[nn][r] * scale;
            }
        }

        // ---- online softmax (half-wave shuffle reductions) ----------------
#pragma unroll
        for (int r = 0; r < 8; ++r) {
            float tm = fmaxf(fmaxf(sac[0][r], sac[1][r]),
                             fmaxf(sac[2][r], sac[3][r]));
            tm = hmax16(tm);
            const float mnew  = fmaxf(mrow[r], tm);
            const bool  dead  = (mnew == -INFINITY);
            const float alpha = dead ? 1.0f : __expf(mrow[r] - mnew);
            float psum = 0.0f;
#pragma unroll
            for (int nn = 0; nn < 4; ++nn) {
                const float p = dead ? 0.0f : __expf(sac[nn][r] - mnew);
                psum += p;
                lP[wave][half * 8 + r][nn * 16 + lmod] = (bf16)p;
            }
            psum    = hsum16(psum);
            lrow[r] = lrow[r] * alpha + psum;
            mrow[r] = mnew;
#pragma unroll
            for (int nt = 0; nt < 4; ++nt) oacc[nt][r] *= alpha;
        }

        // ---- O += P V  (per-wave LDS ops are in-order; no barrier needed) -
#pragma unroll
        for (int kk = 0; kk < 2; ++kk) {
            AFrag pa;
            const bf16* prow = &lP[wave][lmod][kk * 32];
            pa.h[0] = *(const v8bf*)(prow + half * 16);
            pa.h[1] = *(const v8bf*)(prow + half * 16 + 8);
#pragma unroll
            for (int nt = 0; nt < 4; ++nt) {
                AFrag vv;
                vv.v = *(const v16bf*)&lV[nt * 16 + lmod][kk * 32 + half * 16];
                oacc[nt] = WMMA_BF16(pa.v, vv.v, oacc[nt]);
            }
        }
        __syncthreads();
    }

    // ---- normalize and store y (b, t, h*64+d) bf16 ------------------------
#pragma unroll
    for (int nt = 0; nt < 4; ++nt) {
#pragma unroll
        for (int r = 0; r < 8; ++r) {
            const int   t = qblk * 64 + wave * 16 + half * 8 + r;
            const float l = lrow[r];
            const float o = (l > 0.0f) ? oacc[nt][r] / l : 0.0f;
            yb[((size_t)b * Tc + t) * Cc + h * Dc + nt * 16 + lmod] = (bf16)o;
        }
    }
}

// ---------------------------------------------------------------------------
// Kernel 3: output projection.  out[m, n] = y[m,:] . Wo[n,:] + bo[n]   (f32)
// ---------------------------------------------------------------------------
__global__ __launch_bounds__(128) void out_proj_kernel(
    const bf16* __restrict__ y, const float* __restrict__ Wo,
    const float* __restrict__ bo, float* __restrict__ out) {
    __shared__ bf16 lA[64][32];
    __shared__ bf16 lB[64][32];

    const int mblk = blockIdx.x * 64;
    const int nblk = blockIdx.y * 64;

    const int tid  = threadIdx.x;
    const int wave = tid >> 5;
    const int lane = tid & 31;
    const int half = lane >> 4;
    const int lmod = lane & 15;

    v8f acc[4];
#pragma unroll
    for (int i = 0; i < 4; ++i) acc[i] = vzero8();

    const int row  = tid >> 1;
    const int kseg = (tid & 1) * 16;

    for (int k0 = 0; k0 < Cc; k0 += 32) {
        {
            const bf16*  ys = y + (size_t)(mblk + row) * Cc + k0 + kseg;
            const float* ws = Wo + (size_t)(nblk + row) * Cc + k0 + kseg;
#pragma unroll
            for (int j = 0; j < 16; ++j) lA[row][kseg + j] = ys[j];
#pragma unroll
            for (int j = 0; j < 16; ++j) lB[row][kseg + j] = (bf16)ws[j];
        }
        __syncthreads();

        AFrag a;
        a.h[0] = *(const v8bf*)&lA[wave * 16 + lmod][half * 16];
        a.h[1] = *(const v8bf*)&lA[wave * 16 + lmod][half * 16 + 8];
#pragma unroll
        for (int nt = 0; nt < 4; ++nt) {
            AFrag b;
            b.v = *(const v16bf*)&lB[nt * 16 + lmod][half * 16];
            acc[nt] = WMMA_BF16(a.v, b.v, acc[nt]);
        }
        __syncthreads();
    }

#pragma unroll
    for (int nt = 0; nt < 4; ++nt) {
        const int   n  = nblk + nt * 16 + lmod;
        const float bv = bo[n];
#pragma unroll
        for (int r = 0; r < 8; ++r) {
            const int m = mblk + wave * 16 + half * 8 + r;
            out[(size_t)m * Cc + n] = acc[nt][r] + bv;
        }
    }
}

// ---------------------------------------------------------------------------
extern "C" void kernel_launch(void* const* d_in, const int* in_sizes, int n_in,
                              void* d_out, int out_size, void* d_ws,
                              size_t ws_size, hipStream_t stream) {
    (void)in_sizes; (void)n_in; (void)out_size; (void)ws_size;

    const float*         x     = (const float*)d_in[0];
    const unsigned char* dmask = (const unsigned char*)d_in[1];  // (B,1,T,T) bool
    const unsigned char* cmask = (const unsigned char*)d_in[2];  // (1,1,T,T) bool
    const float* Wq = (const float*)d_in[3];
    const float* bq = (const float*)d_in[4];
    const float* Wk = (const float*)d_in[5];
    const float* bk = (const float*)d_in[6];
    const float* Wv = (const float*)d_in[7];
    const float* bv = (const float*)d_in[8];
    const float* Wo = (const float*)d_in[9];
    const float* bo = (const float*)d_in[10];

    const size_t nqkv = (size_t)Bc * Hc * Tc * Dc;  // 4M elements
    bf16* qb = (bf16*)d_ws;
    bf16* kb = qb + nqkv;
    bf16* vb = kb + nqkv;
    bf16* yb = vb + nqkv;

    dim3 blk(128);
    dim3 g1((Bc * Tc) / 64, Cc / 64, 3);
    qkv_proj_kernel<<<g1, blk, 0, stream>>>(x, Wq, bq, Wk, bk, Wv, bv, qb, kb, vb);

    dim3 g2(Tc / 64, Bc * Hc);
    attn_kernel<<<g2, blk, 0, stream>>>(qb, kb, vb, cmask, dmask, yb);

    dim3 g3((Bc * Tc) / 64, Cc / 64);
    out_proj_kernel<<<g3, blk, 0, stream>>>(yb, Wo, bo, (float*)d_out);
}
